// MultiHeadAttention_68796786147593
// MI455X (gfx1250) — matrix operand order repaired and checked by reading, not compiled, not run on hardware
//
#include <hip/hip_runtime.h>

typedef _Float16 h16;
typedef __attribute__((ext_vector_type(8)))  _Float16 v8h;
typedef __attribute__((ext_vector_type(16))) _Float16 v16h;
typedef __attribute__((ext_vector_type(8)))  float    v8f;

#define B_   2
#define S_   2048
#define D_   1024
#define H_   16
#define DH_  64

static __device__ __forceinline__ v16h make_frag(v8h lo, v8h hi) {
  union { v16h v; v8h h[2]; } u;
  u.h[0] = lo; u.h[1] = hi;
  return u.v;
}

// gfx1250 async copy global->LDS (ASYNCcnt path, no VGPR round trip).
// vdst VGPR holds the per-lane LDS byte offset (low 32 bits of generic LDS ptr).
static __device__ __forceinline__ void async_load_b128(const h16* gptr, h16* lptr) {
  unsigned lds = (unsigned)(size_t)lptr;
  unsigned long long ga = (unsigned long long)(size_t)gptr;
  asm volatile("global_load_async_to_lds_b128 %0, %1, off"
               :: "v"(lds), "v"(ga) : "memory");
}
static __device__ __forceinline__ void wait_async() {
  asm volatile("s_wait_asynccnt 0x0" ::: "memory");
}

// ---------------------------------------------------------------- conversions
__global__ void k_f32_to_f16(const float* __restrict__ src, h16* __restrict__ dst, int n) {
  int i = blockIdx.x * blockDim.x + threadIdx.x;
  int stride = gridDim.x * blockDim.x;
  for (; i < n; i += stride) dst[i] = (h16)src[i];
}

// W[k][n] (1024x1024 f32, row-major) -> Wt[n][k] (f16)
__global__ void k_w_transpose_f16(const float* __restrict__ src, h16* __restrict__ dst) {
  int i = blockIdx.x * blockDim.x + threadIdx.x;
  int n = i >> 10, k = i & 1023;
  dst[(size_t)n * 1024 + k] = (h16)src[(size_t)k * 1024 + n];
}

// ---------------------------------------------------------------- WMMA GEMM
// C[M=4096, N=1024] = A[4096,1024](f16,row-major) x Bt[N][K](f16) + bias
// block tile 256x64, 8 waves, wave tile 32x64 (8 WMMAs / k-step / wave)
// explicitly 2-stage software pipeline: async global->LDS into one buffer
// while WMMAs consume the other; buffer indices are compile-time constants.
#define GK  1024
#define BM  256
#define BN  64
#define KS  32
#define LDA 40   // 32 halves + 8 pad (16B-aligned b128 LDS ops)

__global__ __launch_bounds__(256)
void k_gemm_f16(const h16* __restrict__ A, const h16* __restrict__ Bt,
                const float* __restrict__ bias, h16* __restrict__ outh,
                float* __restrict__ outf, int mode) {
  __shared__ __align__(16) h16 sA[2][BM * LDA];   // 2 x 20 KB
  __shared__ __align__(16) h16 sB[2][BN * LDA];   // 2 x 5 KB

  const int t    = threadIdx.x;
  const int lane = t & 31;
  const int wave = t >> 5;
  const int r15  = lane & 15;
  const int hsel = lane >> 4;
  const int m0   = blockIdx.y * BM;
  const int n0   = blockIdx.x * BN;

  v8f acc[2][4];
#pragma unroll
  for (int i = 0; i < 2; ++i)
#pragma unroll
    for (int j = 0; j < 4; ++j)
#pragma unroll
      for (int e = 0; e < 8; ++e) acc[i][j][e] = 0.0f;

  // async-stage one 256x32 A tile + 64x32 B tile into buffer `buf`
  auto stage = [&](int buf, int kk) {
#pragma unroll
    for (int it = 0; it < 4; ++it) {               // 1024 chunks of 8 halves
      int c = (it << 8) | t;                        // uniform trip count
      int row = c >> 2, off = (c & 3) * 8;
      async_load_b128(&A[(size_t)(m0 + row) * GK + kk + off],
                      &sA[buf][row * LDA + off]);
    }
    int row = t >> 2, off = (t & 3) * 8;
    async_load_b128(&Bt[(size_t)(n0 + row) * GK + kk + off],
                    &sB[buf][row * LDA + off]);
  };

  // 8 WMMAs against buffer `buf`
  auto compute = [&](int buf) {
    v16h af[2];
#pragma unroll
    for (int i = 0; i < 2; ++i) {
      const h16* ar = &sA[buf][(32 * wave + 16 * i + r15) * LDA];
      af[i] = make_frag(*(const v8h*)&ar[hsel * 8],
                        *(const v8h*)&ar[hsel * 8 + 16]);
    }
#pragma unroll
    for (int j = 0; j < 4; ++j) {
      const h16* br = &sB[buf][(16 * j + r15) * LDA];
      v16h bf = make_frag(*(const v8h*)&br[hsel * 8],
                          *(const v8h*)&br[hsel * 8 + 16]);
#pragma unroll
      for (int i = 0; i < 2; ++i)
        acc[i][j] = __builtin_amdgcn_wmma_f32_16x16x32_f16(
            false, af[i], false, bf, (short)0, acc[i][j], false, false);
    }
  };

  // prologue
  stage(0, 0);
  wait_async();
  __syncthreads();

  // main pipeline, unrolled by 2 so buffer indices are constant
  for (int k0 = 0; k0 < GK; k0 += 2 * KS) {
    // phase A: stage buf1 (k0+KS always < GK since GK/KS is even), compute buf0
    stage(1, k0 + KS);
    compute(0);
    wait_async();
    __syncthreads();
    // phase B: stage buf0 with k0+2*KS (skip on last iteration), compute buf1
    if (k0 + 2 * KS < GK) stage(0, k0 + 2 * KS);
    compute(1);
    wait_async();
    __syncthreads();
  }

  // epilogue; C/D layout: lane L, vgpr r -> row = r + 8*(L>=16), col = L&15
#pragma unroll
  for (int j = 0; j < 4; ++j) {
    int col = n0 + 16 * j + r15;
    float bv = bias[col];
#pragma unroll
    for (int i = 0; i < 2; ++i)
#pragma unroll
      for (int r = 0; r < 8; ++r) {
        int row = m0 + 32 * wave + 16 * i + 8 * hsel + r;
        float val = acc[i][j][r] + bv;
        if (mode == 0) {
          int b = row >> 11, s = row & 2047;
          int h = col >> 6,  d = col & 63;
          outh[(((size_t)b * H_ + h) * S_ + s) * DH_ + d] = (h16)val;
        } else {
          outf[(size_t)row * 1024 + col] = val;
        }
      }
  }
}

// ---------------------------------------------------------------- flash attention
// grid (S/64, B*H), 128 threads (4 waves). Each wave owns 16 query rows.
#define LDK 72   // 64 halves + 8 pad

__global__ __launch_bounds__(128)
void k_flash_attn(const h16* __restrict__ Q, const h16* __restrict__ K,
                  const h16* __restrict__ V, h16* __restrict__ AO) {
  __shared__ __align__(16) h16 sK[64 * LDK];         // K tile, row-major [sk][d]
  __shared__ __align__(16) h16 sV[64 * LDK];         // V tile, transposed [d][sk]
  __shared__ __align__(16) h16 sP[4 * 16 * LDK];     // per-wave P tile [row][sk]

  const int t    = threadIdx.x;
  const int lane = t & 31;
  const int wave = t >> 5;
  const int r15  = lane & 15;
  const int hsel = lane >> 4;
  const int bh   = blockIdx.y;               // b*H + h
  const int q0   = blockIdx.x * 64;
  const size_t headBase = (size_t)bh * S_ * DH_;

  // Q fragments for this wave's 16-row strip, resident across the whole KV loop
  const h16* qrow = &Q[headBase + (size_t)(q0 + 16 * wave + r15) * DH_];
  v16h qf0 = make_frag(*(const v8h*)&qrow[hsel * 8],
                       *(const v8h*)&qrow[hsel * 8 + 16]);
  v16h qf1 = make_frag(*(const v8h*)&qrow[32 + hsel * 8],
                       *(const v8h*)&qrow[32 + hsel * 8 + 16]);

  float mrow[8], lrow[8];
  v8f o[4];
#pragma unroll
  for (int r = 0; r < 8; ++r) { mrow[r] = -3.0e38f; lrow[r] = 0.0f; }
#pragma unroll
  for (int j = 0; j < 4; ++j)
#pragma unroll
    for (int e = 0; e < 8; ++e) o[j][e] = 0.0f;

  const int ntiles = blockIdx.x + 1;   // causal: only KV tiles at or left of diag
  for (int tile = 0; tile < ntiles; ++tile) {
    const int k0 = tile * 64;
    __syncthreads();
    // stage K via async global->LDS; V transposed via VGPRs
#pragma unroll
    for (int it = 0; it < 4; ++it) {
      int c = (it << 7) | t;                 // uniform trip count
      int row = c >> 3, off = (c & 7) * 8;
      async_load_b128(&K[headBase + (size_t)(k0 + row) * DH_ + off],
                      &sK[row * LDK + off]);
      v8h vv = *(const v8h*)&V[headBase + (size_t)(k0 + row) * DH_ + off];
#pragma unroll
      for (int e = 0; e < 8; ++e) sV[(off + e) * LDK + row] = vv[e];
    }
    // prefetch next KV tile into caches while we compute this one
    if (tile + 1 < ntiles) {
      int prow = t >> 1, poff = (t & 1) * 32;
      __builtin_prefetch(&K[headBase + (size_t)(k0 + 64 + prow) * DH_ + poff], 0, 1);
      __builtin_prefetch(&V[headBase + (size_t)(k0 + 64 + prow) * DH_ + poff], 0, 1);
    }
    wait_async();
    __syncthreads();

    // S = Q * K^T  (B column n == K row n; contiguous over Dh in sK)
    v8f s[4];
#pragma unroll
    for (int j = 0; j < 4; ++j) {
#pragma unroll
      for (int e = 0; e < 8; ++e) s[j][e] = 0.0f;
      const h16* krow = &sK[(16 * j + r15) * LDK];
      v16h b0 = make_frag(*(const v8h*)&krow[hsel * 8],
                          *(const v8h*)&krow[hsel * 8 + 16]);
      v16h b1 = make_frag(*(const v8h*)&krow[32 + hsel * 8],
                          *(const v8h*)&krow[32 + hsel * 8 + 16]);
      s[j] = __builtin_amdgcn_wmma_f32_16x16x32_f16(
          false, qf0, false, b0, (short)0, s[j], false, false);
      s[j] = __builtin_amdgcn_wmma_f32_16x16x32_f16(
          false, qf1, false, b1, (short)0, s[j], false, false);
    }

    // scale + causal mask (analytic; only bites on the diagonal tile)
    const int qbase = q0 + 16 * wave + 8 * hsel;
#pragma unroll
    for (int j = 0; j < 4; ++j) {
      int kcol = k0 + 16 * j + r15;
#pragma unroll
      for (int r = 0; r < 8; ++r) {
        float sc = s[j][r] * 0.125f;   // 1/sqrt(64)
        s[j][r] = (kcol > qbase + r) ? -1.0e30f : sc;
      }
    }

    // online softmax: rows live in vgpr index + lane-half; reduce over 16 lanes
#pragma unroll
    for (int r = 0; r < 8; ++r) {
      float tm = fmaxf(fmaxf(s[0][r], s[1][r]), fmaxf(s[2][r], s[3][r]));
#pragma unroll
      for (int msk = 1; msk <= 8; msk <<= 1)
        tm = fmaxf(tm, __shfl_xor(tm, msk, 32));
      float newm = fmaxf(mrow[r], tm);
      float ts = 0.0f;
#pragma unroll
      for (int j = 0; j < 4; ++j) {
        float p = __expf(s[j][r] - newm);
        s[j][r] = p;
        ts += p;
      }
#pragma unroll
      for (int msk = 1; msk <= 8; msk <<= 1)
        ts += __shfl_xor(ts, msk, 32);
      float alpha = __expf(mrow[r] - newm);
      lrow[r] = lrow[r] * alpha + ts;
      mrow[r] = newm;
#pragma unroll
      for (int j = 0; j < 4; ++j) o[j][r] *= alpha;
    }

    // C-layout -> A-layout via per-wave LDS round trip
    h16* pw = &sP[wave * 16 * LDK];
#pragma unroll
    for (int j = 0; j < 4; ++j)
#pragma unroll
      for (int r = 0; r < 8; ++r)
        pw[(8 * hsel + r) * LDK + 16 * j + r15] = (h16)s[j][r];

    v16h pf0 = make_frag(*(const v8h*)&pw[r15 * LDK + hsel * 8],
                         *(const v8h*)&pw[r15 * LDK + hsel * 8 + 16]);
    v16h pf1 = make_frag(*(const v8h*)&pw[r15 * LDK + 32 + hsel * 8],
                         *(const v8h*)&pw[r15 * LDK + 32 + hsel * 8 + 16]);

    // O += P * V   (B column n == Dh index; contiguous over sk in transposed sV)
#pragma unroll
    for (int j = 0; j < 4; ++j) {
      const h16* vrow = &sV[(16 * j + r15) * LDK];
      v16h vb0 = make_frag(*(const v8h*)&vrow[hsel * 8],
                           *(const v8h*)&vrow[hsel * 8 + 16]);
      v16h vb1 = make_frag(*(const v8h*)&vrow[32 + hsel * 8],
                           *(const v8h*)&vrow[32 + hsel * 8 + 16]);
      o[j] = __builtin_amdgcn_wmma_f32_16x16x32_f16(
          false, pf0, false, vb0, (short)0, o[j], false, false);
      o[j] = __builtin_amdgcn_wmma_f32_16x16x32_f16(
          false, pf1, false, vb1, (short)0, o[j], false, false);
    }
  }

  // finalize + store f16 into AO[B][S][H][Dh]
  const int b = bh >> 4, h = bh & 15;
#pragma unroll
  for (int r = 0; r < 8; ++r) {
    float inv = 1.0f / lrow[r];
    int srow = q0 + 16 * wave + 8 * hsel + r;
    size_t rowbase = (((size_t)b * S_ + srow) * H_ + h) * DH_;
#pragma unroll
    for (int j = 0; j < 4; ++j)
      AO[rowbase + 16 * j + r15] = (h16)(o[j][r] * inv);
  }
}

// ---------------------------------------------------------------- launch
extern "C" void kernel_launch(void* const* d_in, const int* in_sizes, int n_in,
                              void* d_out, int out_size, void* d_ws, size_t ws_size,
                              hipStream_t stream) {
  (void)in_sizes; (void)n_in; (void)out_size; (void)ws_size;

  const float* x_q = (const float*)d_in[0];
  const float* x_k = (const float*)d_in[1];
  const float* x_v = (const float*)d_in[2];
  // d_in[3] = mask_mat (bool): causal, computed analytically in-kernel
  const float* Wq = (const float*)d_in[4];
  const float* bq = (const float*)d_in[5];
  const float* Wk = (const float*)d_in[6];
  const float* bk = (const float*)d_in[7];
  const float* Wv = (const float*)d_in[8];
  const float* bv = (const float*)d_in[9];
  const float* Wo = (const float*)d_in[10];
  const float* bo = (const float*)d_in[11];
  float* out = (float*)d_out;

  const size_t NX = (size_t)B_ * S_ * D_;   // 4,194,304
  const size_t NW = (size_t)D_ * D_;        // 1,048,576

  // workspace layout (f16), 32M halves = 64 MB total
  h16* ws  = (h16*)d_ws;
  h16* Xq  = ws;
  h16* Xk  = Xq  + NX;
  h16* Xv  = Xk  + NX;
  h16* Wqt = Xv  + NX;
  h16* Wkt = Wqt + NW;
  h16* Wvt = Wkt + NW;
  h16* Wot = Wvt + NW;
  h16* Qh  = Wot + NW;   // [B,H,S,Dh]
  h16* Kh  = Qh  + NX;
  h16* Vh  = Kh  + NX;
  h16* AOh = Vh  + NX;   // [B,S,H,Dh]

  k_f32_to_f16<<<4096, 256, 0, stream>>>(x_q, Xq, (int)NX);
  k_f32_to_f16<<<4096, 256, 0, stream>>>(x_k, Xk, (int)NX);
  k_f32_to_f16<<<4096, 256, 0, stream>>>(x_v, Xv, (int)NX);
  k_w_transpose_f16<<<(int)(NW / 256), 256, 0, stream>>>(Wq, Wqt);
  k_w_transpose_f16<<<(int)(NW / 256), 256, 0, stream>>>(Wk, Wkt);
  k_w_transpose_f16<<<(int)(NW / 256), 256, 0, stream>>>(Wv, Wvt);
  k_w_transpose_f16<<<(int)(NW / 256), 256, 0, stream>>>(Wo, Wot);

  dim3 gg(1024 / BN, 4096 / BM);    // (Ntiles, Mtiles) = (16, 16)
  k_gemm_f16<<<gg, 256, 0, stream>>>(Xq, Wqt, bq, Qh, nullptr, 0);
  k_gemm_f16<<<gg, 256, 0, stream>>>(Xk, Wkt, bk, Kh, nullptr, 0);
  k_gemm_f16<<<gg, 256, 0, stream>>>(Xv, Wvt, bv, Vh, nullptr, 0);

  dim3 ga(S_ / 64, B_ * H_);        // (q tiles, heads)
  k_flash_attn<<<ga, 128, 0, stream>>>(Qh, Kh, Vh, AOh);

  k_gemm_f16<<<gg, 256, 0, stream>>>(AOh, Wot, bo, nullptr, out, 1);
}
